// DRBKernel_20701742367268
// MI455X (gfx1250) — matrix operand order repaired and checked by reading, not compiled, not run on hardware
//
#include <hip/hip_runtime.h>
#include <hip/hip_bf16.h>

typedef __bf16 bf16_t;
typedef __attribute__((ext_vector_type(16))) __bf16 v16bf;
typedef __attribute__((ext_vector_type(8)))  __bf16 v8bf;
typedef __attribute__((ext_vector_type(8)))  float  v8f;

__device__ __forceinline__ float leakyf(float v) { return v >= 0.f ? v : 0.1f * v; }

// ---------------------------------------------------------------------------
// Weight repack: f32 (Cout, Cin, 3, 3)  ->  bf16 [tap][cin_chunk32][cout][32]
// so the WMMA A-stage is a contiguous coalesced copy.
// ---------------------------------------------------------------------------
__global__ void repack_w(const float* __restrict__ w, bf16_t* __restrict__ wp,
                         int Cin, int Cout, int taps) {
    int total = Cout * Cin * taps;
    int i = blockIdx.x * blockDim.x + threadIdx.x;
    if (i >= total) return;
    int j  = i & 31;
    int r  = i >> 5;
    int co = r % Cout;
    int q  = r / Cout;
    int nCC = Cin >> 5;
    int cc = q % nCC;
    int t  = q / nCC;
    int ci = cc * 32 + j;
    wp[i] = (bf16_t)w[((size_t)co * Cin + ci) * taps + t];
}

// ---------------------------------------------------------------------------
// Implicit-GEMM 3x3 conv via v_wmma_f32_16x16x32_bf16, double-buffered LDS.
// Block: 256 threads = 8 waves. Tile: 64 Cout x 128 pixels.
// Wave wv: coSub = wv%4 (16 couts), pSub = wv/4 (64 pixels = 4 accums).
// Per K-chunk per wave: 2 A-frag + 8 B-frag ds_load_b128 -> 4 WMMAs (A reuse x4),
// one barrier per chunk; staging addresses advance incrementally (ptr += 32*P),
// tap-boundary recompute only every nCC chunks (uniform branch).
// Requires: Cin % 32 == 0, Cout % 64 == 0, (H*W) % 128 == 0. Zero padding.
// ---------------------------------------------------------------------------
#define LDS_STRIDE 40  // 32 bf16 payload + 8 pad per row

__global__ __launch_bounds__(256) void wmma_conv3(
    const float* __restrict__ in, const bf16_t* __restrict__ wp,
    const float* __restrict__ bias, float* __restrict__ out,
    int Cin, int Cout, int H, int W, int act)
{
    __shared__ __align__(16) bf16_t Alds[2][64  * LDS_STRIDE];
    __shared__ __align__(16) bf16_t Blds[2][128 * LDS_STRIDE];

    const int P      = H * W;
    const int tid    = threadIdx.x;
    const int pBase  = blockIdx.x * 128;
    const int coBase = blockIdx.y * 64;
    const int nCC    = Cin >> 5;
    const int total  = 9 * nCC;

    const int lane  = tid & 31;
    const int wv    = tid >> 5;
    const int coSub = wv & 3;
    const int pSub  = wv >> 2;

    v8f acc[4] = {{}, {}, {}, {}};

    // staging coordinates (B tile): thread -> pixel (tid%128), k-group (tid/128)*16
    const int pl = tid & 127;
    const int jg = (tid >> 7) * 16;     // 0 or 16
    const int p  = pBase + pl;
    const int py = p / W, px = p % W;

    // fragment LDS offsets (per ISA VGPR layout tables)
    const int half   = lane >> 4;
    const int aOff   = (coSub * 16 + (lane & 15)) * LDS_STRIDE + half * 8;
    const int prOff  = (pSub * 64 + (lane & 15)) * LDS_STRIDE + half * 8;
    const int aStOff = (tid >> 2) * LDS_STRIDE + (tid & 3) * 8;

    // staging state (advanced incrementally per chunk)
    const bf16_t* aSrc = wp + (size_t)coBase * 32 + (size_t)tid * 8;
    const float*  bptr;
    bool          binb;
    auto setTap = [&](int t) {
        const int dy = t / 3 - 1, dx = t % 3 - 1;
        const int yy = py + dy, xx = px + dx;
        binb = (yy >= 0) && (yy < H) && (xx >= 0) && (xx < W);
        const int yyc = yy < 0 ? 0 : (yy >= H ? H - 1 : yy);
        const int xxc = xx < 0 ? 0 : (xx >= W ? W - 1 : xx);
        bptr = in + (size_t)jg * P + (size_t)yyc * W + xxc;
    };
    setTap(0);
    int tS = 0, ccS = 0;

    auto stage = [&](int bi) {
        // ---- A: 64x32 bf16, contiguous, one b128 per thread ----
        *(v8bf*)&Alds[bi][aStOff] = *(const v8bf*)aSrc;
        aSrc += (size_t)Cout * 32;
        // ---- B: branch-free im2col, clause of 16 loads then pack ----
        float f[16];
        #pragma unroll
        for (int j = 0; j < 16; ++j) f[j] = bptr[(size_t)j * P];
        #pragma unroll
        for (int sub = 0; sub < 2; ++sub) {
            v8bf pk;
            #pragma unroll
            for (int j = 0; j < 8; ++j)
                pk[j] = (bf16_t)(binb ? f[sub * 8 + j] : 0.f);
            *(v8bf*)&Blds[bi][pl * LDS_STRIDE + jg + sub * 8] = pk;
        }
        // advance chunk state: next cin chunk, or next tap (uniform, rare)
        bptr += (size_t)32 * P;
        if (++ccS == nCC) {
            ccS = 0;
            ++tS;
            if (tS < 9) setTap(tS);
        }
    };

    stage(0);
    for (int q = 0; q < total; ++q) {
        __syncthreads();
        const int bi = q & 1;

        v8bf alo = *(const v8bf*)&Alds[bi][aOff];
        v8bf ahi = *(const v8bf*)&Alds[bi][aOff + 16];
        v16bf a;
        #pragma unroll
        for (int e = 0; e < 8; ++e) { a[e] = alo[e]; a[e + 8] = ahi[e]; }

        #pragma unroll
        for (int i = 0; i < 4; ++i) {
            v8bf bl = *(const v8bf*)&Blds[bi][prOff + i * 16 * LDS_STRIDE];
            v8bf bh = *(const v8bf*)&Blds[bi][prOff + i * 16 * LDS_STRIDE + 16];
            v16bf b;
            #pragma unroll
            for (int e = 0; e < 8; ++e) { b[e] = bl[e]; b[e + 8] = bh[e]; }
            acc[i] = __builtin_amdgcn_wmma_f32_16x16x32_bf16(
                false, a, false, b, (short)0, acc[i], false, false);
        }
        if (q + 1 < total) stage((q + 1) & 1);
    }

    // ---- epilogue: D layout VGPR r -> M = r + 8*(lane/16), N = lane%16 ----
    const int n    = lane & 15;
    const int mAdd = 8 * (lane >> 4);
    #pragma unroll
    for (int i = 0; i < 4; ++i) {
        const int pix = pBase + pSub * 64 + i * 16 + n;
        #pragma unroll
        for (int r = 0; r < 8; ++r) {
            int co = coBase + coSub * 16 + r + mAdd;
            float v = acc[i][r] + bias[co];
            if (act) v = leakyf(v);
            out[(size_t)co * P + pix] = v;
        }
    }
}

// ---------------------------------------------------------------------------
// Direct conv (Cin=3 stems, 1x1 heads). Zero padding.
// ---------------------------------------------------------------------------
__global__ void conv_direct(const float* __restrict__ in, const float* __restrict__ w,
                            const float* __restrict__ bias, float* __restrict__ out,
                            int Cin, int Cout, int H, int W, int ksz, int act)
{
    int P = H * W;
    int idx = blockIdx.x * blockDim.x + threadIdx.x;
    if (idx >= Cout * P) return;
    int pp = idx % P, co = idx / P;
    int y = pp / W, x = pp % W;
    int pad = (ksz - 1) / 2;
    float acc = bias[co];
    for (int ci = 0; ci < Cin; ++ci) {
        const float* ip = in + (size_t)ci * P;
        const float* wl = w + ((size_t)co * Cin + ci) * ksz * ksz;
        for (int ky = 0; ky < ksz; ++ky) {
            int yy = y + ky - pad;
            if (yy < 0 || yy >= H) continue;
            for (int kx = 0; kx < ksz; ++kx) {
                int xx = x + kx - pad;
                if (xx < 0 || xx >= W) continue;
                acc += ip[(size_t)yy * W + xx] * wl[ky * ksz + kx];
            }
        }
    }
    if (act) acc = leakyf(acc);
    out[idx] = acc;
}

// 8x8 area pool: (3,H,W) -> (3,H/8,W/8)
__global__ void avgpool8_k(const float* __restrict__ in, float* __restrict__ out,
                           int H, int W) {
    int Ho = H / 8, Wo = W / 8, Po = Ho * Wo;
    int idx = blockIdx.x * blockDim.x + threadIdx.x;
    if (idx >= 3 * Po) return;
    int pp = idx % Po, c = idx / Po;
    int oy = pp / Wo, ox = pp % Wo;
    float s = 0.f;
    for (int dy = 0; dy < 8; ++dy)
        for (int dx = 0; dx < 8; ++dx)
            s += in[(size_t)c * H * W + (size_t)(oy * 8 + dy) * W + (ox * 8 + dx)];
    out[idx] = s * (1.f / 64.f);
}

// nearest 2x upsample: (3,H,W) -> (3,2H,2W)
__global__ void upsample2_k(const float* __restrict__ in, float* __restrict__ out,
                            int H, int W) {
    int Ho = H * 2, Wo = W * 2, Po = Ho * Wo;
    int idx = blockIdx.x * blockDim.x + threadIdx.x;
    if (idx >= 3 * Po) return;
    int pp = idx % Po, c = idx / Po;
    int oy = pp / Wo, ox = pp % Wo;
    out[idx] = in[(size_t)c * H * W + (size_t)(oy >> 1) * W + (ox >> 1)];
}

// est = x + FAC(x, k7x7, replicate pad) + r ; optional clip to [-1,1]
__global__ void fac_est_k(const float* __restrict__ x, const float* __restrict__ k,
                          const float* __restrict__ r, float* __restrict__ out,
                          int H, int W, int doClip) {
    int P = H * W;
    int idx = blockIdx.x * blockDim.x + threadIdx.x;
    if (idx >= 3 * P) return;
    int pp = idx % P, c = idx / P;
    int y = pp / W, xc = pp % W;
    float acc = x[idx] + r[idx];
    for (int a = 0; a < 7; ++a) {
        int yy = y + a - 3; yy = yy < 0 ? 0 : (yy >= H ? H - 1 : yy);
        for (int b = 0; b < 7; ++b) {
            int xx = xc + b - 3; xx = xx < 0 ? 0 : (xx >= W ? W - 1 : xx);
            acc += x[(size_t)c * P + (size_t)yy * W + xx] * k[(size_t)(a * 7 + b) * P + pp];
        }
    }
    if (doClip) acc = fminf(fmaxf(acc, -1.f), 1.f);
    out[idx] = acc;
}

// ---------------------------------------------------------------------------
extern "C" void kernel_launch(void* const* d_in, const int* in_sizes, int n_in,
                              void* d_out, int out_size, void* d_ws, size_t ws_size,
                              hipStream_t stream) {
    (void)in_sizes; (void)n_in; (void)out_size; (void)ws_size;

    const float* Cimg = (const float*)d_in[0];
    const float* fs0  = (const float*)d_in[1];
    const float* fs1  = (const float*)d_in[2];
    const float* fs2  = (const float*)d_in[3];
    const float* fs3  = (const float*)d_in[4];

    // params flattened jax-pytree style: dict keys sorted alphabetically,
    // each spec = 3 layers of (w, b).
    enum { D1 = 0, D2, D4, D8, U0K, U0R, U1K, U1R, U2K, U2R, U3K, U3R };
    const float* Wp[12][3]; const float* Bp[12][3];
    int pi = 5;
    for (int s = 0; s < 12; ++s)
        for (int l = 0; l < 3; ++l) {
            Wp[s][l] = (const float*)d_in[pi++];
            Bp[s][l] = (const float*)d_in[pi++];
        }

    const int P1 = 384 * 384, P2 = 192 * 192, P4 = 96 * 96, P8 = 48 * 48;

    // workspace bump allocator
    char* wsb = (char*)d_ws;
    size_t off = 0;
    auto alloc = [&](size_t bytes) -> void* {
        void* pr = wsb + off;
        off = (off + bytes + 255) & ~(size_t)255;
        return pr;
    };
    float*  feat = (float*)alloc((size_t)128 * P1 * 4);  // also >= 1024*P8, 512*P4, 256*P2
    float*  tA   = (float*)alloc((size_t)128 * P1 * 4);
    float*  tB   = (float*)alloc((size_t)128 * P1 * 4);
    float*  kb   = (float*)alloc((size_t)49 * P1 * 4);
    float*  xb   = (float*)alloc((size_t)3 * P1 * 4);
    float*  rb   = (float*)alloc((size_t)3 * P1 * 4);
    float*  eb   = (float*)alloc((size_t)3 * P1 * 4);
    bf16_t* wpb  = (bf16_t*)alloc((size_t)4718592 * 2);  // largest layer 1024*512*9

    auto wmmaConv = [&](const float* in, const float* w, const float* b, float* out,
                        int Cin, int Cout, int H, int W, int act) {
        int total = Cout * Cin * 9;
        repack_w<<<(total + 255) / 256, 256, 0, stream>>>(w, wpb, Cin, Cout, 9);
        dim3 grid((H * W) / 128, Cout / 64);
        wmma_conv3<<<grid, 256, 0, stream>>>(in, wpb, b, out, Cin, Cout, H, W, act);
    };
    auto dirConv = [&](const float* in, const float* w, const float* b, float* out,
                       int Cin, int Cout, int H, int W, int ksz, int act) {
        long total = (long)Cout * H * W;
        conv_direct<<<(int)((total + 255) / 256), 256, 0, stream>>>(
            in, w, b, out, Cin, Cout, H, W, ksz, act);
    };

    // =============== scale 8 (48x48) ===============
    avgpool8_k<<<(3 * P8 + 255) / 256, 256, 0, stream>>>(Cimg, xb, 384, 384);
    dirConv(xb, Wp[D8][0], Bp[D8][0], tA, 3, 128, 48, 48, 3, 1);
    wmmaConv(tA, Wp[D8][1], Bp[D8][1], tB, 128, 256, 48, 48, 1);
    wmmaConv(tB, Wp[D8][2], Bp[D8][2], feat + (size_t)512 * P8, 256, 512, 48, 48, 1);
    hipMemcpyAsync(feat, fs0, (size_t)512 * P8 * 4, hipMemcpyDeviceToDevice, stream);
    wmmaConv(feat, Wp[U3K][0], Bp[U3K][0], tA, 1024, 512, 48, 48, 1);
    wmmaConv(tA, Wp[U3K][1], Bp[U3K][1], tB, 512, 256, 48, 48, 1);
    dirConv(tB, Wp[U3K][2], Bp[U3K][2], kb, 256, 49, 48, 48, 1, 0);
    wmmaConv(feat, Wp[U3R][0], Bp[U3R][0], tA, 1024, 512, 48, 48, 1);
    wmmaConv(tA, Wp[U3R][1], Bp[U3R][1], tB, 512, 128, 48, 48, 1);
    dirConv(tB, Wp[U3R][2], Bp[U3R][2], rb, 128, 3, 48, 48, 1, 1);
    fac_est_k<<<(3 * P8 + 255) / 256, 256, 0, stream>>>(xb, kb, rb, eb, 48, 48, 0);
    upsample2_k<<<(3 * P4 + 255) / 256, 256, 0, stream>>>(eb, xb, 48, 48);

    // =============== scale 4 (96x96) ===============
    dirConv(xb, Wp[D4][0], Bp[D4][0], tA, 3, 128, 96, 96, 3, 1);
    wmmaConv(tA, Wp[D4][1], Bp[D4][1], tB, 128, 256, 96, 96, 1);
    wmmaConv(tB, Wp[D4][2], Bp[D4][2], feat + (size_t)256 * P4, 256, 256, 96, 96, 1);
    hipMemcpyAsync(feat, fs1, (size_t)256 * P4 * 4, hipMemcpyDeviceToDevice, stream);
    wmmaConv(feat, Wp[U2K][0], Bp[U2K][0], tA, 512, 256, 96, 96, 1);
    wmmaConv(tA, Wp[U2K][1], Bp[U2K][1], tB, 256, 256, 96, 96, 1);
    dirConv(tB, Wp[U2K][2], Bp[U2K][2], kb, 256, 49, 96, 96, 1, 0);
    wmmaConv(feat, Wp[U2R][0], Bp[U2R][0], tA, 512, 256, 96, 96, 1);
    wmmaConv(tA, Wp[U2R][1], Bp[U2R][1], tB, 256, 128, 96, 96, 1);
    dirConv(tB, Wp[U2R][2], Bp[U2R][2], rb, 128, 3, 96, 96, 1, 1);
    fac_est_k<<<(3 * P4 + 255) / 256, 256, 0, stream>>>(xb, kb, rb, eb, 96, 96, 0);
    upsample2_k<<<(3 * P2 + 255) / 256, 256, 0, stream>>>(eb, xb, 96, 96);

    // =============== scale 2 (192x192) ===============
    dirConv(xb, Wp[D2][0], Bp[D2][0], tA, 3, 128, 192, 192, 3, 1);
    wmmaConv(tA, Wp[D2][1], Bp[D2][1], tB, 128, 128, 192, 192, 1);
    wmmaConv(tB, Wp[D2][2], Bp[D2][2], feat + (size_t)128 * P2, 128, 128, 192, 192, 1);
    hipMemcpyAsync(feat, fs2, (size_t)128 * P2 * 4, hipMemcpyDeviceToDevice, stream);
    wmmaConv(feat, Wp[U1K][0], Bp[U1K][0], tA, 256, 128, 192, 192, 1);
    wmmaConv(tA, Wp[U1K][1], Bp[U1K][1], tB, 128, 128, 192, 192, 1);
    dirConv(tB, Wp[U1K][2], Bp[U1K][2], kb, 128, 49, 192, 192, 1, 0);
    wmmaConv(feat, Wp[U1R][0], Bp[U1R][0], tA, 256, 128, 192, 192, 1);
    wmmaConv(tA, Wp[U1R][1], Bp[U1R][1], tB, 128, 128, 192, 192, 1);
    dirConv(tB, Wp[U1R][2], Bp[U1R][2], rb, 128, 3, 192, 192, 1, 1);
    fac_est_k<<<(3 * P2 + 255) / 256, 256, 0, stream>>>(xb, kb, rb, eb, 192, 192, 0);
    upsample2_k<<<(3 * P1 + 255) / 256, 256, 0, stream>>>(eb, xb, 192, 192);

    // =============== scale 1 (384x384) ===============
    dirConv(xb, Wp[D1][0], Bp[D1][0], tA, 3, 128, 384, 384, 3, 1);
    wmmaConv(tA, Wp[D1][1], Bp[D1][1], tB, 128, 128, 384, 384, 1);
    wmmaConv(tB, Wp[D1][2], Bp[D1][2], feat + (size_t)64 * P1, 128, 64, 384, 384, 1);
    hipMemcpyAsync(feat, fs3, (size_t)64 * P1 * 4, hipMemcpyDeviceToDevice, stream);
    wmmaConv(feat, Wp[U0K][0], Bp[U0K][0], tA, 128, 128, 384, 384, 1);
    wmmaConv(tA, Wp[U0K][1], Bp[U0K][1], tB, 128, 128, 384, 384, 1);
    dirConv(tB, Wp[U0K][2], Bp[U0K][2], kb, 128, 49, 384, 384, 1, 0);
    wmmaConv(feat, Wp[U0R][0], Bp[U0R][0], tA, 128, 128, 384, 384, 1);
    wmmaConv(tA, Wp[U0R][1], Bp[U0R][1], tB, 128, 128, 384, 384, 1);
    dirConv(tB, Wp[U0R][2], Bp[U0R][2], rb, 128, 3, 384, 384, 1, 1);
    fac_est_k<<<(3 * P1 + 255) / 256, 256, 0, stream>>>(xb, kb, rb, (float*)d_out,
                                                        384, 384, 1);
}
